// ImprovedCGCNN_14319420965091
// MI455X (gfx1250) — matrix operand order repaired and checked by reading, not compiled, not run on hardware
//
#include <hip/hip_runtime.h>
#include <hip/hip_bf16.h>

typedef __attribute__((ext_vector_type(16))) _Float16 v16h;
typedef __attribute__((ext_vector_type(8)))  _Float16 v8h;
typedef __attribute__((ext_vector_type(8)))  float    v8f;
typedef __attribute__((ext_vector_type(4))) unsigned int u32x4;
typedef __attribute__((ext_vector_type(8))) int          i32x8;
typedef __attribute__((ext_vector_type(4))) int          i32x4;

#define F_DIM 128
#define ZDIM  297
#define NTILE 10            // K padded to 320 = 10 * 32
#define NCONV 6
#define NG    128
#define XDIM  98
#define EDGE_DIM 41

// Packed B image (per layer,kTile): 2 mats x 8 jTiles x (32 lanes x 32B, swizzled)
#define JT_BYTES   1088     // 4 lane-groups * (8*32B + 16B pad) = 4*272
#define MAT_BYTES  (8 * JT_BYTES)          // 8704
#define TILE_BYTES (2 * MAT_BYTES)         // 17408
#define TILE_ELEMS (TILE_BYTES / 8)        // 2176 8-byte elements
#define JT_HALF    (JT_BYTES / 2)          // 544
#define MAT_HALF   (MAT_BYTES / 2)         // 4352
#define TILE_HALF  (TILE_BYTES / 2)        // 8704

// ---------------------------------------------------------------------------
// Prepack conv weights into the contiguous TDM image.
// Within a (layer,t) tile: [mat][j][8-lane group: 8 lanes x 16 halves + 8 pad]
// lane byte offset = lane*32 + (lane>>3)*16  (bank-conflict-free swizzle)
// B fragment layout (32x16 f16): lanes 0-15 -> col n=j*16+lane, K = 32t + q;
//                                lanes 16-31 -> col n-16,       K = 32t + 16 + q.
// ---------------------------------------------------------------------------
__global__ void prepack_weights(const float* __restrict__ Wf,
                                const float* __restrict__ Ws,
                                _Float16* __restrict__ Bp) {
  const int total = NCONV * NTILE * TILE_HALF;
  int idx = blockIdx.x * 256 + threadIdx.x;
  if (idx >= total) return;
  int lt   = idx / TILE_HALF;                 // layer*10 + t
  int rem  = idx - lt * TILE_HALF;
  int layer = lt / NTILE;
  int t     = lt - layer * NTILE;
  int mat  = rem / MAT_HALF;  int rem2 = rem - mat * MAT_HALF;
  int j    = rem2 / JT_HALF;  int rem3 = rem2 - j * JT_HALF;
  int g    = rem3 / 136;      int rem4 = rem3 - g * 136;
  float v = 0.0f;
  if (rem4 < 128) {
    int lane = g * 8 + (rem4 >> 4);
    int q    = rem4 & 15;
    int hi   = lane >> 4;
    int n    = j * 16 + (lane & 15);
    int k    = 32 * t + 16 * hi + q;
    if (k < ZDIM) {
      const float* W = (mat ? Ws : Wf) + (size_t)layer * ZDIM * F_DIM;
      v = W[(size_t)k * F_DIM + n];
    }
  }
  Bp[idx] = (_Float16)v;
}

// edge_attr [E,41] fp32 -> padded [E,64] f16 (cols 41..63 zero)
__global__ void convert_edge_attr(const float* __restrict__ ea,
                                  _Float16* __restrict__ ea16, long E) {
  long idx = (long)blockIdx.x * 256 + threadIdx.x;
  if (idx >= E * 64) return;
  long e = idx >> 6;
  int  c = (int)(idx & 63);
  float v = (c < EDGE_DIM) ? ea[e * EDGE_DIM + c] : 0.0f;
  ea16[idx] = (_Float16)v;
}

// h = x @ embed_W + embed_b ; also emit f16 copy
__global__ void embed_kernel(const float* __restrict__ x,
                             const float* __restrict__ W,
                             const float* __restrict__ b,
                             float* __restrict__ h,
                             _Float16* __restrict__ h16, long N) {
  __shared__ float xr[XDIM];
  long node = blockIdx.x;
  int  n    = threadIdx.x;
  if (node >= N) return;
  if (n < XDIM) xr[n] = x[node * XDIM + n];
  __syncthreads();
  float acc = b[n];
  #pragma unroll 7
  for (int k = 0; k < XDIM; k++) acc += xr[k] * W[(size_t)k * F_DIM + n];
  h[node * F_DIM + n]   = acc;
  h16[node * F_DIM + n] = (_Float16)acc;
}

// ---------------------------------------------------------------------------
// TDM: DMA one 17408-byte weight tile from global into LDS (1-D, 8-byte elems)
// D# built per CDNA5 ISA 8.3/8.4 bitfields. 6-arg builtin (clang-23 flavor).
// ---------------------------------------------------------------------------
__device__ __forceinline__ void tdm_issue(unsigned long long gaddr, unsigned ldsaddr) {
  u32x4 g0;
  g0[0] = 1u;                                            // count=1, user descriptor
  g0[1] = ldsaddr;                                       // lds_addr
  g0[2] = (unsigned)(gaddr & 0xFFFFFFFFu);               // global_addr[31:0]
  g0[3] = (unsigned)((gaddr >> 32) & 0x01FFFFFFu)        // global_addr[56:32]
        | 0x80000000u;                                   // type=2 ("image")
  i32x8 g1;
  g1[0] = 0x00030000;                                    // data_size=3 (8B), mask=0
  g1[1] = (int)((unsigned)TILE_ELEMS << 16);             // tensor_dim0[15:0] @ [63:48]
  g1[2] = (int)(((unsigned)TILE_ELEMS >> 16) | (1u << 16)); // dim0[31:16], tensor_dim1=1
  g1[3] = (int)((unsigned)TILE_ELEMS << 16);             // tile_dim0 @ [127:112]
  g1[4] = 1;                                             // tile_dim1=1, tile_dim2=0
  g1[5] = TILE_ELEMS;                                    // tensor_dim0_stride lo32
  g1[6] = (int)((unsigned)TILE_ELEMS << 16);             // dim1_stride lo16 @ [223:208]
  g1[7] = 0;
  i32x4 gz4 = {0, 0, 0, 0};
  i32x8 gz8 = {0, 0, 0, 0, 0, 0, 0, 0};
  __builtin_amdgcn_tensor_load_to_lds(g0, g1, gz4, gz4, gz8, 0);
}

// ---------------------------------------------------------------------------
// Fused edge GEMM + gate + scatter-add.
// Per wave: 16 edges x 128 outputs; A = [h16[dst] | h16[src] | ea16[e]] (K->320).
// B tiles TDM-staged into LDS, double buffered. 160 WMMA per wave.
// ---------------------------------------------------------------------------
__global__ __launch_bounds__(256)
void edge_gemm_scatter(const _Float16* __restrict__ h16,
                       const _Float16* __restrict__ ea16,
                       const _Float16* __restrict__ Bp,
                       const int* __restrict__ edge_index,
                       const float* __restrict__ conv_bf,
                       const float* __restrict__ conv_bs,
                       float* __restrict__ agg,
                       long E, int layer) {
  __shared__ __align__(16) _Float16 ldsbuf[2 * TILE_HALF];

  const int tid  = threadIdx.x;
  const int lane = tid & 31;
  const int wave = tid >> 5;
  const int r  = lane & 15;
  const int hi = lane >> 4;
  const long eBase = (long)blockIdx.x * 128 + wave * 16;
  long e  = eBase + r;
  long ec = (e < E) ? e : (E - 1);
  const int srcN = edge_index[ec];       // edge_index[0] = src (j)
  const int dstN = edge_index[E + ec];   // edge_index[1] = dst (i)

  const unsigned ldsBase = (unsigned)(uintptr_t)(&ldsbuf[0]);  // flat low 32 = LDS offset
  const unsigned long long gBase =
      (unsigned long long)(uintptr_t)Bp + (unsigned long long)(layer * NTILE) * TILE_BYTES;
  const int laneHalf = lane * 16 + (lane >> 3) * 8;            // swizzled lane offset (halves)

  v8f zero = {0.f,0.f,0.f,0.f,0.f,0.f,0.f,0.f};
  v8f accF[8], accS[8];
  #pragma unroll
  for (int j = 0; j < 8; j++) { accF[j] = zero; accS[j] = zero; }

  #pragma unroll
  for (int t = 0; t < NTILE; t++) {
    const int cur = t & 1;
    __syncthreads();               // prior readers of the buffer we are about to fill are done
    if (wave == 0) {
      if (t == 0)
        tdm_issue(gBase, ldsBase);
      if (t + 1 < NTILE)
        tdm_issue(gBase + (unsigned long long)(t + 1) * TILE_BYTES,
                  ldsBase + (unsigned)(1 - cur) * TILE_BYTES);
      if (t + 1 < NTILE) __builtin_amdgcn_s_wait_tensorcnt(1);  // tile t complete (in-order)
      else               __builtin_amdgcn_s_wait_tensorcnt(0);
    }
    __syncthreads();               // tile t visible to all waves

    // A fragment (contiguous, matches packed B): lane<16 -> K {0..15}+32t,
    //                                             lane>=16 -> K {16..31}+32t
    int k0 = 32 * t + 16 * hi;
    int k1 = k0 + 8;
    const _Float16* p0 = (k0 < 128) ? h16 + (size_t)dstN * F_DIM + k0
                       : (k0 < 256) ? h16 + (size_t)srcN * F_DIM + (k0 - 128)
                                    : ea16 + (size_t)ec * 64 + (k0 - 256);
    const _Float16* p1 = (k1 < 128) ? h16 + (size_t)dstN * F_DIM + k1
                       : (k1 < 256) ? h16 + (size_t)srcN * F_DIM + (k1 - 128)
                                    : ea16 + (size_t)ec * 64 + (k1 - 256);
    v8h c0 = *(const v8h*)p0;
    v8h c1 = *(const v8h*)p1;
    v16h a = __builtin_shufflevector(c0, c1, 0,1,2,3,4,5,6,7,8,9,10,11,12,13,14,15);

    const _Float16* lbase = ldsbuf + cur * TILE_HALF;
    #pragma unroll
    for (int j = 0; j < 8; j++) {
      const _Float16* bfp = lbase + j * JT_HALF + laneHalf;
      v8h f0 = *(const v8h*)bfp;
      v8h f1 = *(const v8h*)(bfp + 8);
      v16h bF = __builtin_shufflevector(f0, f1, 0,1,2,3,4,5,6,7,8,9,10,11,12,13,14,15);
      accF[j] = __builtin_amdgcn_wmma_f32_16x16x32_f16(
          false, a, false, bF, (short)0, accF[j], false, false);
      const _Float16* bsp = lbase + MAT_HALF + j * JT_HALF + laneHalf;
      v8h s0 = *(const v8h*)bsp;
      v8h s1 = *(const v8h*)(bsp + 8);
      v16h bS = __builtin_shufflevector(s0, s1, 0,1,2,3,4,5,6,7,8,9,10,11,12,13,14,15);
      accS[j] = __builtin_amdgcn_wmma_f32_16x16x32_f16(
          false, a, false, bS, (short)0, accS[j], false, false);
    }
  }

  // Epilogue. C layout: VGPR rr, lane<16 -> (M=rr, N=lane); lane>=16 -> (M=rr+8, N=lane-16)
  int dstR[8];
  #pragma unroll
  for (int rr = 0; rr < 8; rr++) {
    long e2 = eBase + rr + 8 * hi;
    dstR[rr] = (e2 < E) ? edge_index[E + e2] : -1;
  }
  #pragma unroll
  for (int j = 0; j < 8; j++) {
    int n = j * 16 + r;
    float bfv = conv_bf[layer * F_DIM + n];
    float bsv = conv_bs[layer * F_DIM + n];
    #pragma unroll
    for (int rr = 0; rr < 8; rr++) {
      if (dstR[rr] >= 0) {
        float xf = accF[j][rr] + bfv;
        float xs = accS[j][rr] + bsv;
        float sig = 1.0f / (1.0f + __expf(-xf));
        float sp  = (xs > 15.0f) ? xs : __logf(1.0f + __expf(xs));
        atomicAdd(agg + (size_t)dstR[rr] * F_DIM + n, sig * sp);
      }
    }
  }
}

// BN(agg) + residual + BN + optional residual; re-emit f16; zero agg
__global__ void node_update(float* __restrict__ h, _Float16* __restrict__ h16,
                            float* __restrict__ agg,
                            const float* __restrict__ g_in, const float* __restrict__ b_in,
                            const float* __restrict__ m_in, const float* __restrict__ v_in,
                            const float* __restrict__ g_out, const float* __restrict__ b_out,
                            const float* __restrict__ m_out, const float* __restrict__ v_out,
                            long N, int layer) {
  long idx = (long)blockIdx.x * 256 + threadIdx.x;
  if (idx >= N * F_DIM) return;
  int n = (int)(idx & (F_DIM - 1));
  int o = layer * F_DIM + n;
  float a = agg[idx];
  agg[idx] = 0.0f;
  float t1  = (a - m_in[o]) * (g_in[o] * rsqrtf(v_in[o] + 1e-5f)) + b_in[o];
  float hin = h[idx];
  float h1  = t1 + hin;
  float h2  = (h1 - m_out[o]) * (g_out[o] * rsqrtf(v_out[o] + 1e-5f)) + b_out[o];
  if (layer > 0) h2 += hin;
  h[idx]   = h2;
  h16[idx] = (_Float16)h2;
}

__device__ __forceinline__ unsigned enc_f32(float f) {
  int ib = __float_as_int(f);
  return (ib < 0) ? ~(unsigned)ib : ((unsigned)ib | 0x80000000u);
}

__global__ void pool_accum(const float* __restrict__ h, const int* __restrict__ batch,
                           float* __restrict__ psum, int* __restrict__ pcnt,
                           unsigned* __restrict__ pmax, long N) {
  long idx = (long)blockIdx.x * 256 + threadIdx.x;
  if (idx >= N * F_DIM) return;
  long node = idx >> 7;
  int  n    = (int)(idx & (F_DIM - 1));
  int  g    = batch[node];
  float v   = h[idx];
  atomicAdd(psum + (size_t)g * F_DIM + n, v);
  atomicMax(pmax + (size_t)g * F_DIM + n, enc_f32(v));
  if (n == 0) atomicAdd(pcnt + g, 1);
}

__global__ void pool_finalize(const float* __restrict__ psum, const int* __restrict__ pcnt,
                              const unsigned* __restrict__ pmax, float* __restrict__ xp) {
  int idx = blockIdx.x * 256 + threadIdx.x;
  if (idx >= NG * 256) return;
  int g = idx >> 8;
  int c = idx & 255;
  float out;
  if (c < F_DIM) {
    out = psum[g * F_DIM + c] / fmaxf((float)pcnt[g], 1.0f);
  } else {
    unsigned u = pmax[g * F_DIM + (c - F_DIM)];
    out = 0.0f;
    if (u != 0u) {
      int ib = (u & 0x80000000u) ? (int)(u & 0x7fffffffu) : ~(int)u;
      float mv = __int_as_float(ib);
      out = (__builtin_isfinite(mv)) ? mv : 0.0f;
    }
  }
  xp[idx] = out;
}

// act: 0 none, 1 relu, 2 silu
__global__ void dense_kernel(const float* __restrict__ in, const float* __restrict__ W,
                             const float* __restrict__ b, float* __restrict__ out,
                             int Brows, int K, int O, int act) {
  int idx = blockIdx.x * 256 + threadIdx.x;
  if (idx >= Brows * O) return;
  int row = idx / O, o = idx - row * O;
  const float* ir = in + (size_t)row * K;
  float acc = b[o];
  for (int k = 0; k < K; k++) acc += ir[k] * W[(size_t)k * O + o];
  if (act == 1)      acc = fmaxf(acc, 0.0f);
  else if (act == 2) acc = acc / (1.0f + __expf(-acc));
  out[idx] = acc;
}

// ---------------------------------------------------------------------------
static inline size_t alignUp(size_t x) { return (x + 255) & ~(size_t)255; }

extern "C" void kernel_launch(void* const* d_in, const int* in_sizes, int n_in,
                              void* d_out, int out_size, void* d_ws, size_t ws_size,
                              hipStream_t stream) {
  const float* x          = (const float*)d_in[0];
  const int*   edge_index = (const int*)  d_in[1];
  const float* edge_attr  = (const float*)d_in[2];
  const int*   batch      = (const int*)  d_in[3];
  const float* embed_W    = (const float*)d_in[4];
  const float* embed_b    = (const float*)d_in[5];
  const float* conv_Wf    = (const float*)d_in[6];
  const float* conv_bf    = (const float*)d_in[7];
  const float* conv_Ws    = (const float*)d_in[8];
  const float* conv_bs    = (const float*)d_in[9];
  const float* bn_in_g    = (const float*)d_in[10];
  const float* bn_in_b    = (const float*)d_in[11];
  const float* bn_in_m    = (const float*)d_in[12];
  const float* bn_in_v    = (const float*)d_in[13];
  const float* bn_out_g   = (const float*)d_in[14];
  const float* bn_out_b   = (const float*)d_in[15];
  const float* bn_out_m   = (const float*)d_in[16];
  const float* bn_out_v   = (const float*)d_in[17];
  const float* pool_W     = (const float*)d_in[18];
  const float* pool_b     = (const float*)d_in[19];
  const float* f1_W       = (const float*)d_in[20];
  const float* f1_b       = (const float*)d_in[21];
  const float* f2_W       = (const float*)d_in[22];
  const float* f2_b       = (const float*)d_in[23];
  const float* f3_W       = (const float*)d_in[24];
  const float* f3_b       = (const float*)d_in[25];
  const float* d1_W       = (const float*)d_in[26];
  const float* d1_b       = (const float*)d_in[27];
  const float* d2_W       = (const float*)d_in[28];
  const float* d2_b       = (const float*)d_in[29];
  float* outp = (float*)d_out;

  const long N = in_sizes[0] / XDIM;
  const long E = in_sizes[1] / 2;

  // workspace carve-up
  char* w = (char*)d_ws;
  size_t off = 0;
  float*    h     = (float*)(w + off);    off = alignUp(off + (size_t)N * F_DIM * 4);
  _Float16* h16   = (_Float16*)(w + off); off = alignUp(off + (size_t)N * F_DIM * 2);
  float*    agg   = (float*)(w + off);    off = alignUp(off + (size_t)N * F_DIM * 4);
  _Float16* ea16  = (_Float16*)(w + off); off = alignUp(off + (size_t)E * 64 * 2);
  _Float16* Bpack = (_Float16*)(w + off); off = alignUp(off + (size_t)NCONV * NTILE * TILE_BYTES);
  float*    psum  = (float*)(w + off);    off = alignUp(off + (size_t)NG * F_DIM * 4);
  int*      pcnt  = (int*)(w + off);      off = alignUp(off + (size_t)NG * 4);
  unsigned* pmax  = (unsigned*)(w + off); off = alignUp(off + (size_t)NG * F_DIM * 4);
  float*    xpool = (float*)(w + off);    off = alignUp(off + (size_t)NG * 256 * 4);
  float*    embB  = (float*)(w + off);    off = alignUp(off + (size_t)NG * F_DIM * 4);
  float*    buf1  = (float*)(w + off);    off = alignUp(off + (size_t)NG * 256 * 4);
  float*    buf2  = (float*)(w + off);    off = alignUp(off + (size_t)NG * F_DIM * 4);
  float*    buf3  = (float*)(w + off);    off = alignUp(off + (size_t)NG * 256 * 4);
  (void)ws_size; (void)n_in; (void)out_size;

  // per-call prep: weight prepack + edge_attr f16
  {
    int total = NCONV * NTILE * TILE_HALF;
    prepack_weights<<<(total + 255) / 256, 256, 0, stream>>>(conv_Wf, conv_Ws, Bpack);
  }
  {
    long total = E * 64;
    convert_edge_attr<<<(int)((total + 255) / 256), 256, 0, stream>>>(edge_attr, ea16, E);
  }

  // embedding
  embed_kernel<<<(int)N, F_DIM, 0, stream>>>(x, embed_W, embed_b, h, h16, N);

  // zero accumulators
  (void)hipMemsetAsync(agg, 0, (size_t)N * F_DIM * 4, stream);
  (void)hipMemsetAsync(psum, 0, (size_t)NG * F_DIM * 4, stream);
  (void)hipMemsetAsync(pcnt, 0, (size_t)NG * 4, stream);
  (void)hipMemsetAsync(pmax, 0, (size_t)NG * F_DIM * 4, stream);

  // conv layers
  const int edgeBlocks = (int)((E + 127) / 128);
  const int nodeBlocks = (int)((N * F_DIM + 255) / 256);
  for (int layer = 0; layer < NCONV; layer++) {
    edge_gemm_scatter<<<edgeBlocks, 256, 0, stream>>>(
        h16, ea16, Bpack, edge_index, conv_bf, conv_bs, agg, E, layer);
    node_update<<<nodeBlocks, 256, 0, stream>>>(
        h, h16, agg,
        bn_in_g, bn_in_b, bn_in_m, bn_in_v,
        bn_out_g, bn_out_b, bn_out_m, bn_out_v, N, layer);
  }

  // pooling
  pool_accum<<<nodeBlocks, 256, 0, stream>>>(h, batch, psum, pcnt, pmax, N);
  pool_finalize<<<(NG * 256 + 255) / 256, 256, 0, stream>>>(psum, pcnt, pmax, xpool);

  // heads
  dense_kernel<<<(NG * 128 + 255) / 256, 256, 0, stream>>>(xpool, pool_W, pool_b, embB, NG, 256, 128, 1);
  dense_kernel<<<(NG * 256 + 255) / 256, 256, 0, stream>>>(embB, f1_W, f1_b, buf1, NG, 128, 256, 2);
  dense_kernel<<<(NG * 128 + 255) / 256, 256, 0, stream>>>(buf1, f2_W, f2_b, buf2, NG, 256, 128, 2);
  dense_kernel<<<(NG * 1 + 255) / 256, 256, 0, stream>>>(buf2, f3_W, f3_b, outp, NG, 128, 1, 0);
  dense_kernel<<<(NG * 256 + 255) / 256, 256, 0, stream>>>(embB, d1_W, d1_b, buf3, NG, 128, 256, 2);
  dense_kernel<<<(NG * 384 + 255) / 256, 256, 0, stream>>>(buf3, d2_W, d2_b, outp + NG, NG, 256, 384, 0);
}
